// ProjectedAdaptiveLogSoftmax_40072044871962
// MI455X (gfx1250) — matrix operand order repaired and checked by reading, not compiled, not run on hardware
//
#include <hip/hip_runtime.h>
#include <hip/hip_bf16.h>

// ---------------- problem constants ----------------
#define DDIM      1024
#define NROWS     4096
#define NTOKEN    50000
#define SHORTLIST 20000
#define CUT1      40000

#define KSTEP     32               // WMMA K for bf16
#define KSTEPS    (DDIM / KSTEP)   // 32 k-steps
#define COLS_PER_BLOCK 128         // 8 waves * 16 cols
#define HEAD_BLOCKS  157           // ceil(20000/128)
#define TAIL1_BLOCKS 157           // ceil(20000/128)
#define TAIL2_BLOCKS 79            // ceil(10000/128)
#define TOTAL_BLOCKS (HEAD_BLOCKS + TAIL1_BLOCKS + TAIL2_BLOCKS)  // 393
#define ROW_TILES (NROWS / 16)     // 256
#define PF_DEPTH  4                // A-fragment software pipeline depth

typedef __bf16 bf16_t;
typedef __attribute__((ext_vector_type(16))) __bf16 v16bf;
typedef __attribute__((ext_vector_type(8)))  __bf16 v8bf;
typedef __attribute__((ext_vector_type(8)))  float  v8f;
typedef __attribute__((ext_vector_type(4)))  float  v4f;   // clang vector (nontemporal-load OK)

#define NEG_INF (-__builtin_inff())

// fp32 -> bf16 round-to-nearest-even via bit manipulation (robust to compile).
__device__ __forceinline__ unsigned short f2bf_bits(float f) {
    union { float f; unsigned int u; } v; v.f = f;
    unsigned int r = v.u + 0x7fffu + ((v.u >> 16) & 1u);
    return (unsigned short)(r >> 16);
}
__device__ __forceinline__ bf16_t f2bf(float f) {
    union { unsigned short u; bf16_t h; } o; o.u = f2bf_bits(f);
    return o.h;
}

// online-softmax pair combine: (M,S) <- (M,S) (+) (m,s)
__device__ __forceinline__ void lse_combine(float& M, float& S, float m, float s) {
    float M2 = fmaxf(M, m);
    float a = (M == NEG_INF) ? 0.f : S * __expf(M - M2);
    float b = (m == NEG_INF) ? 0.f : s * __expf(m - M2);
    M = M2; S = a + b;
}

// ---------------- kernel 1: hidden fp32 -> bf16 ----------------
__global__ void cvt_hidden_kernel(const float4* __restrict__ h, ushort4* __restrict__ o) {
    int i = blockIdx.x * blockDim.x + threadIdx.x;   // exactly NROWS*DDIM/4 threads
    float4 f = h[i];
    ushort4 r;
    r.x = f2bf_bits(f.x); r.y = f2bf_bits(f.y);
    r.z = f2bf_bits(f.z); r.w = f2bf_bits(f.w);
    o[i] = r;
}

// ---------------- kernel 2: weight-stationary WMMA GEMM + per-block LSE partials ----
// grid = TOTAL_BLOCKS blocks of 256 threads (8 waves). Wave w owns 16 vocab cols.
// Per wave: 16 cols x 1024 K of weights live in 32 resident v16bf B-fragments
// (weights stream from HBM exactly once, non-temporal), then sweep all 256
// row-tiles of hidden (bf16, L2-resident) with a 4-deep A-fragment pipeline.
__global__ __launch_bounds__(256)
void gemm_lse_kernel(const bf16_t* __restrict__ hb,      // [NROWS][DDIM] bf16
                     const float*  __restrict__ W,       // [NTOKEN][DDIM]
                     const float*  __restrict__ bias,    // [NTOKEN]
                     float2*       __restrict__ part)    // [TOTAL_BLOCKS][NROWS]
{
    __shared__ float2 lds[8 * 16];

    const int b = blockIdx.x;
    int colBase, colEnd;
    if (b < HEAD_BLOCKS)                      { colBase = b * COLS_PER_BLOCK;                              colEnd = SHORTLIST; }
    else if (b < HEAD_BLOCKS + TAIL1_BLOCKS)  { colBase = SHORTLIST + (b - HEAD_BLOCKS) * COLS_PER_BLOCK;  colEnd = CUT1; }
    else                                      { colBase = CUT1 + (b - HEAD_BLOCKS - TAIL1_BLOCKS) * COLS_PER_BLOCK; colEnd = NTOKEN; }

    const int w    = threadIdx.x >> 5;
    const int lane = threadIdx.x & 31;
    const int col0 = colBase + w * 16;
    const bool validWave = (col0 < colEnd);    // colEnd % 16 == 0 -> wave-uniform validity
    const int  n   = lane & 15;                 // column within tile / row within tile
    const int  kbB = (lane < 16) ? 0 : 16;      // B fragment K-base for this lane half
    const int  kbA = (lane < 16) ? 0 : 16;      // A fragment K-base for this lane half

    // ---- resident B slab: 16 cols x 1024 K, bf16, 32 fragments (256 VGPRs) ----
    v16bf bfrag[KSTEPS];
    float biasl = 0.f;
    if (validWave) {
        const float* wrow = W + (size_t)(col0 + n) * DDIM;
        #pragma unroll
        for (int ks = 0; ks < KSTEPS; ++ks) {
            const v4f* p = (const v4f*)(wrow + ks * KSTEP + kbB);
            v4f f[4];
            #pragma unroll
            for (int j = 0; j < 4; ++j)              // single-use weight stream:
                f[j] = __builtin_nontemporal_load(p + j);  // don't evict hidden from L2
            v16bf t;
            #pragma unroll
            for (int e = 0; e < 16; ++e)
                t[e] = f2bf(f[e >> 2][e & 3]);
            bfrag[ks] = t;
        }
        biasl = bias[col0 + n];
    }

    // ---- sweep all row tiles of hidden ----
    for (int rt = 0; rt < ROW_TILES; ++rt) {
        if (validWave) {
            const bf16_t* arow = hb + (size_t)(rt * 16 + n) * DDIM;

            // software pipeline: keep PF_DEPTH A-fragments in flight
            v8bf alo[PF_DEPTH], ahi[PF_DEPTH];
            #pragma unroll
            for (int p = 0; p < PF_DEPTH; ++p) {
                alo[p] = *(const v8bf*)(arow + p * KSTEP + kbA);
                ahi[p] = *(const v8bf*)(arow + p * KSTEP + 8 + kbA);
            }

            v8f acc = {0.f, 0.f, 0.f, 0.f, 0.f, 0.f, 0.f, 0.f};
            #pragma unroll
            for (int ks = 0; ks < KSTEPS; ++ks) {
                v16bf afrag = __builtin_shufflevector(alo[ks % PF_DEPTH], ahi[ks % PF_DEPTH],
                    0, 1, 2, 3, 4, 5, 6, 7, 8, 9, 10, 11, 12, 13, 14, 15);
                acc = __builtin_amdgcn_wmma_f32_16x16x32_bf16(
                    false, afrag, false, bfrag[ks], (short)0, acc, false, false);
                if (ks + PF_DEPTH < KSTEPS) {
                    alo[ks % PF_DEPTH] = *(const v8bf*)(arow + (ks + PF_DEPTH) * KSTEP + kbA);
                    ahi[ks % PF_DEPTH] = *(const v8bf*)(arow + (ks + PF_DEPTH) * KSTEP + 8 + kbA);
                }
            }

            // per-row (max, sumexp) over this wave's 16 columns.
            // acc[i]: lanes 0-15 -> row rt*16+i, lanes 16-31 -> row rt*16+i+8, col = col0+(lane&15)
            #pragma unroll
            for (int i = 0; i < 8; ++i) {
                float logit = acc[i] + biasl;
                float m = logit;
                m = fmaxf(m, __shfl_xor(m, 8, 16));
                m = fmaxf(m, __shfl_xor(m, 4, 16));
                m = fmaxf(m, __shfl_xor(m, 2, 16));
                m = fmaxf(m, __shfl_xor(m, 1, 16));
                float e = __expf(logit - m);
                e += __shfl_xor(e, 8, 16);
                e += __shfl_xor(e, 4, 16);
                e += __shfl_xor(e, 2, 16);
                e += __shfl_xor(e, 1, 16);
                if ((lane & 15) == 0)
                    lds[w * 16 + i + ((lane >> 4) << 3)] = make_float2(m, e);
            }
        } else {
            if ((lane & 15) == 0) {
                #pragma unroll
                for (int i = 0; i < 8; ++i)
                    lds[w * 16 + i + ((lane >> 4) << 3)] = make_float2(NEG_INF, 0.f);
            }
        }
        __syncthreads();
        if (threadIdx.x < 16) {
            float M = NEG_INF, S = 0.f;
            #pragma unroll
            for (int ww = 0; ww < 8; ++ww) {
                float2 p = lds[ww * 16 + threadIdx.x];
                lse_combine(M, S, p.x, p.y);
            }
            part[(size_t)b * NROWS + rt * 16 + threadIdx.x] = make_float2(M, S);
        }
        __syncthreads();
    }
}

// ---------------- kernel 3: finalize (one wave per row) ----------------
__device__ __forceinline__ float2 seg_lse(const float2* __restrict__ part,
                                          int blkOff, int blkCnt, int row, int lane) {
    float M = NEG_INF, S = 0.f;
    for (int bb = lane; bb < blkCnt; bb += 32) {
        float2 p = part[(size_t)(blkOff + bb) * NROWS + row];
        lse_combine(M, S, p.x, p.y);
    }
    #pragma unroll
    for (int off = 16; off > 0; off >>= 1) {
        float Mo = __shfl_xor(M, off);
        float So = __shfl_xor(S, off);
        lse_combine(M, S, Mo, So);
    }
    return make_float2(M, S);
}

__global__ __launch_bounds__(256)
void finalize_kernel(const float* __restrict__ hidden,
                     const int*   __restrict__ target,
                     const float* __restrict__ W,
                     const float* __restrict__ bias,
                     const float* __restrict__ cw,     // [2][DDIM]
                     const float* __restrict__ cb,     // [2]
                     const float2* __restrict__ part,
                     float*       __restrict__ out)
{
    const int row  = blockIdx.x * 8 + (threadIdx.x >> 5);   // grid sized exactly: NROWS/8 blocks
    const int lane = threadIdx.x & 31;

    int t = target[row];
    if (t < 0) t = 0; if (t >= NTOKEN) t = NTOKEN - 1;

    // fp32 dot products: target logit + the two cluster pseudo-logits
    const float* h  = hidden + (size_t)row * DDIM;
    const float* wt = W + (size_t)t * DDIM;
    float at = 0.f, a0 = 0.f, a1 = 0.f;
    for (int j = lane; j < DDIM; j += 32) {
        float hv = h[j];
        at += hv * wt[j];
        a0 += hv * cw[j];
        a1 += hv * cw[DDIM + j];
    }
    #pragma unroll
    for (int off = 16; off > 0; off >>= 1) {
        at += __shfl_xor(at, off);
        a0 += __shfl_xor(a0, off);
        a1 += __shfl_xor(a1, off);
    }
    const float tl = at + bias[t];
    const float c0 = a0 + cb[0];
    const float c1 = a1 + cb[1];

    // per-segment log-partition
    float2 H  = seg_lse(part, 0, HEAD_BLOCKS, row, lane);
    lse_combine(H.x, H.y, c0, 1.f);       // fold cluster pseudo-tokens into the head
    lse_combine(H.x, H.y, c1, 1.f);
    float2 T1 = seg_lse(part, HEAD_BLOCKS, TAIL1_BLOCKS, row, lane);
    float2 T2 = seg_lse(part, HEAD_BLOCKS + TAIL1_BLOCKS, TAIL2_BLOCKS, row, lane);

    const float logZh = H.x  + __logf(H.y);
    const float logZ1 = T1.x + __logf(T1.y);
    const float logZ2 = T2.x + __logf(T2.y);

    float nll;
    if (t < SHORTLIST)      nll = logZh - tl;
    else if (t < CUT1)      nll = (logZh - c0) + (logZ1 - tl);
    else                    nll = (logZh - c1) + (logZ2 - tl);

    if (lane == 0) out[row] = nll;
}

// ---------------- launch ----------------
extern "C" void kernel_launch(void* const* d_in, const int* in_sizes, int n_in,
                              void* d_out, int out_size, void* d_ws, size_t ws_size,
                              hipStream_t stream) {
    const float* hidden         = (const float*)d_in[0];
    const int*   target         = (const int*)  d_in[1];
    const float* out_weight     = (const float*)d_in[2];
    const float* out_bias       = (const float*)d_in[3];
    const float* cluster_weight = (const float*)d_in[4];
    const float* cluster_bias   = (const float*)d_in[5];
    float* out = (float*)d_out;

    // workspace layout
    bf16_t* hb   = (bf16_t*)d_ws;                                     // 8 MB
    float2* part = (float2*)((char*)d_ws + (size_t)NROWS * DDIM * 2); // ~13 MB

    // 1) hidden fp32 -> bf16  (NROWS*DDIM/4 = 1M float4 threads)
    cvt_hidden_kernel<<<(NROWS * DDIM / 4) / 256, 256, 0, stream>>>(
        (const float4*)hidden, (ushort4*)hb);

    // 2) bf16 WMMA GEMM + per-block online-softmax partials
    gemm_lse_kernel<<<TOTAL_BLOCKS, 256, 0, stream>>>(hb, out_weight, out_bias, part);

    // 3) finalize NLL (one wave per row)
    finalize_kernel<<<NROWS / 8, 256, 0, stream>>>(
        hidden, target, out_weight, out_bias, cluster_weight, cluster_bias, part, out);
}
